// RAFT_6038724018950
// MI455X (gfx1250) — matrix operand order repaired and checked by reading, not compiled, not run on hardware
//
#include <hip/hip_runtime.h>

// ---------------------------------------------------------------------------
// Types for CDNA5 WMMA (wave32)
// ---------------------------------------------------------------------------
typedef __attribute__((ext_vector_type(16))) _Float16 v16h;
typedef __attribute__((ext_vector_type(8)))  float    v8f;

static __device__ __forceinline__ float sigmoidf_(float x) {
    return 1.0f / (1.0f + expf(-x));
}

// ---------------------------------------------------------------------------
// Weight repack: HWIO f32 -> pair-major f16 [Kpad/2][CoutPad][2]
// element (k, n) lives at ((k>>1)*CoutPad + n)*2 + (k&1)
// so a WMMA B-fragment pair (k, k+1) at fixed n is one aligned b32 load.
// ---------------------------------------------------------------------------
__global__ void raft_repack_kernel(const float* __restrict__ w,
                                   _Float16* __restrict__ wp,
                                   int K, int Cout, int Kpad, int CoutPad) {
    int t = blockIdx.x * 256 + threadIdx.x;
    if (t >= Kpad * CoutPad) return;
    int k = t / CoutPad;
    int n = t - k * CoutPad;
    float v = (k < K && n < Cout) ? w[(size_t)k * Cout + n] : 0.0f;
    wp[((size_t)(k >> 1) * CoutPad + n) * 2 + (k & 1)] = (_Float16)v;
}

// ---------------------------------------------------------------------------
// Implicit-GEMM convolution, NHWC f16 in -> NHWC f32 out (+bias).
// Each wave: 16 output pixels x 32 output channels (two 16x16 N-tiles
// sharing one A fragment -> 2x V_WMMA_F32_16X16X32_F16 per K-step).
// A tile staged through LDS (row pad -> conflict-free ds reads); B fragments
// read from pair-major weights with a prefetch of the next K-step.
// ---------------------------------------------------------------------------
#define CONV_WAVES 8

__global__ __launch_bounds__(256) void raft_conv_wmma_kernel(
    const _Float16* __restrict__ x, const _Float16* __restrict__ wp,
    const float* __restrict__ bias, float* __restrict__ y,
    int B, int H, int W, int Cin, int OH, int OW, int Cout,
    int KH, int KW, int stride, int padLo, int Kpad, int CoutPad) {

    __shared__ _Float16 As[16][34];   // 16 pixels x 32 K (+2 pad)

    const int NP      = B * OH * OW;
    const int pixBase = blockIdx.x * 16;
    const int tid  = threadIdx.x;
    const int wave = tid >> 5;
    const int lane = tid & 31;
    const int half = lane >> 4;
    const int mn   = lane & 15;       // row for A-frag, col for B-frag
    // two adjacent 16-channel tiles per wave
    const int cBase0 = (blockIdx.y * CONV_WAVES + wave) * 32;
    const int cBase1 = cBase0 + 16;
    const bool act0  = (cBase0 < CoutPad);
    const bool act1  = (cBase1 < CoutPad);

    const int K   = KH * KW * Cin;
    const int KWC = KW * Cin;

    // Cooperative staging: thread tid fills As[tid>>4][2*(tid&15) .. +1]
    const int sp = tid >> 4;
    const int sk = (tid & 15) * 2;
    int sb = 0, soy = 0, sox = 0;
    bool pvalid = false;
    {
        int gp = pixBase + sp;
        if (gp < NP) {
            pvalid = true;
            sb = gp / (OH * OW);
            int r = gp - sb * OH * OW;
            soy = r / OW;
            sox = r - soy * OW;
        }
    }

    v8f acc0 = {};
    v8f acc1 = {};

    for (int k0 = 0; k0 < Kpad; k0 += 32) {
        // ---- stage A tile ----
        #pragma unroll
        for (int e = 0; e < 2; ++e) {
            int kk = k0 + sk + e;
            _Float16 v = (_Float16)0.0f;
            if (pvalid && kk < K) {
                int kh = kk / KWC;
                int r  = kk - kh * KWC;
                int kw = r / Cin;
                int ci = r - kw * Cin;
                int iy = soy * stride + kh - padLo;
                int ix = sox * stride + kw - padLo;
                if ((unsigned)iy < (unsigned)H && (unsigned)ix < (unsigned)W)
                    v = x[(((size_t)sb * H + iy) * W + ix) * Cin + ci];
            }
            As[sp][sk + e] = v;
        }
        __syncthreads();

        // prefetch next K-step's weight lines into cache (global_prefetch_b8)
        if (act0 && k0 + 32 < Kpad) {
            __builtin_prefetch(
                wp + ((size_t)((k0 + 32) >> 1) * CoutPad + cBase0 + mn) * 2, 0, 3);
        }

        if (act0) {
            v16h af, bf0, bf1;
            #pragma unroll
            for (int vv = 0; vv < 8; ++vv) {
                // A 16x32 f16 layout (matches B): lanes<16 hold K=0..15,
                // lanes>=16 K=16..31.
                int ka = half * 16 + 2 * vv;
                af[2 * vv]     = As[mn][ka];
                af[2 * vv + 1] = As[mn][ka + 1];
                // B 32x16 f16 layout: lanes<16 hold K=0..15, lanes>=16 K=16..31.
                int kb = k0 + half * 16 + 2 * vv;
                const _Float16* bp =
                    wp + ((size_t)(kb >> 1) * CoutPad + (cBase0 + mn)) * 2;
                bf0[2 * vv]     = bp[0];
                bf0[2 * vv + 1] = bp[1];
                bf1[2 * vv]     = act1 ? bp[32] : (_Float16)0.0f;
                bf1[2 * vv + 1] = act1 ? bp[33] : (_Float16)0.0f;
            }
            acc0 = __builtin_amdgcn_wmma_f32_16x16x32_f16(
                false, af, false, bf0, (short)0, acc0, false, false);
            acc1 = __builtin_amdgcn_wmma_f32_16x16x32_f16(
                false, af, false, bf1, (short)0, acc1, false, false);
        }
        __syncthreads();
    }

    if (act0) {
        #pragma unroll
        for (int rr = 0; rr < 8; ++rr) {
            int mm = rr + 8 * half;              // C/D layout: M = r + 8*half
            int gp = pixBase + mm;
            if (gp < NP) {
                int c0 = cBase0 + mn;
                if (c0 < Cout) {
                    float v = acc0[rr];
                    if (bias) v += bias[c0];
                    y[(size_t)gp * Cout + c0] = v;
                }
                int c1 = cBase1 + mn;
                if (act1 && c1 < Cout) {
                    float v = acc1[rr];
                    if (bias) v += bias[c1];
                    y[(size_t)gp * Cout + c1] = v;
                }
            }
        }
    }
}

// ---------------------------------------------------------------------------
// Instance norm over (H,W) per (b,c); f32 in -> f16 out, optional ReLU.
// ---------------------------------------------------------------------------
__global__ __launch_bounds__(256) void raft_inorm_kernel(
    const float* __restrict__ x, const float* __restrict__ g,
    const float* __restrict__ b, _Float16* __restrict__ y,
    int HW, int C, int relu) {
    int bc = blockIdx.x;
    int c  = bc % C;
    int bi = bc / C;
    const float* xp = x + (size_t)bi * HW * C + c;
    __shared__ float s1[256], s2[256];
    float sum = 0.0f, sq = 0.0f;
    for (int i = threadIdx.x; i < HW; i += 256) {
        float v = xp[(size_t)i * C];
        sum += v; sq += v * v;
    }
    s1[threadIdx.x] = sum; s2[threadIdx.x] = sq;
    __syncthreads();
    for (int s = 128; s > 0; s >>= 1) {
        if (threadIdx.x < s) {
            s1[threadIdx.x] += s1[threadIdx.x + s];
            s2[threadIdx.x] += s2[threadIdx.x + s];
        }
        __syncthreads();
    }
    float mean = s1[0] / (float)HW;
    float var  = s2[0] / (float)HW - mean * mean;
    float scale = rsqrtf(var + 1e-3f) * g[c];
    float shift = b[c] - mean * scale;
    _Float16* yp = y + (size_t)bi * HW * C + c;
    for (int i = threadIdx.x; i < HW; i += 256) {
        float v = xp[(size_t)i * C] * scale + shift;
        if (relu) v = fmaxf(v, 0.0f);
        yp[(size_t)i * C] = (_Float16)v;
    }
}

// ---------------------------------------------------------------------------
// Elementwise helpers
// ---------------------------------------------------------------------------
__global__ void raft_f32_to_f16_kernel(const float* __restrict__ x,
                                       _Float16* __restrict__ y,
                                       long n, int relu) {
    long i = (long)blockIdx.x * 256 + threadIdx.x;
    if (i >= n) return;
    float v = x[i];
    if (relu) v = fmaxf(v, 0.0f);
    y[i] = (_Float16)v;
}

__global__ void raft_tanh_kernel(const float* __restrict__ x,
                                 float* __restrict__ h32,
                                 _Float16* __restrict__ h16, long n) {
    long i = (long)blockIdx.x * 256 + threadIdx.x;
    if (i >= n) return;
    float v = tanhf(x[i]);
    h32[i] = v;
    h16[i] = (_Float16)v;
}

__global__ void raft_zsig_kernel(const float* __restrict__ a,
                                 const float* __restrict__ b,
                                 float* __restrict__ z, long n) {
    long i = (long)blockIdx.x * 256 + threadIdx.x;
    if (i >= n) return;
    z[i] = sigmoidf_(a[i] + b[i]);
}

__global__ void raft_rh_kernel(const float* __restrict__ a,
                               const float* __restrict__ b,
                               const float* __restrict__ h,
                               _Float16* __restrict__ rh, long n) {
    long i = (long)blockIdx.x * 256 + threadIdx.x;
    if (i >= n) return;
    rh[i] = (_Float16)(sigmoidf_(a[i] + b[i]) * h[i]);
}

__global__ void raft_hupd_kernel(const float* __restrict__ qa,
                                 const float* __restrict__ qb,
                                 const float* __restrict__ z,
                                 float* __restrict__ h32,
                                 _Float16* __restrict__ h16, long n) {
    long i = (long)blockIdx.x * 256 + threadIdx.x;
    if (i >= n) return;
    float q  = tanhf(qa[i] + qb[i]);
    float zz = z[i];
    float hn = (1.0f - zz) * h32[i] + zz * q;
    h32[i] = hn;
    h16[i] = (_Float16)hn;
}

__global__ void raft_flowadd_kernel(const float* __restrict__ d,
                                    float* __restrict__ flow, long n) {
    long i = (long)blockIdx.x * 256 + threadIdx.x;
    if (i >= n) return;
    flow[i] += d[i];
}

__global__ void raft_mi_kernel(const float* __restrict__ corr,
                               const float* __restrict__ flow,
                               _Float16* __restrict__ mi, int NP) {
    long t = (long)blockIdx.x * 256 + threadIdx.x;
    long tot = (long)NP * 83;
    if (t >= tot) return;
    int p = (int)(t / 83);
    int c = (int)(t - (long)p * 83);
    float v = (c < 81) ? corr[(size_t)p * 81 + c]
                       : flow[(size_t)p * 2 + (c - 81)];
    mi[t] = (_Float16)v;
}

// 81-offset windowed correlation (dot over C, zero outside image)
__global__ void raft_corr_kernel(const float* __restrict__ f1,
                                 const float* __restrict__ f2,
                                 float* __restrict__ corr,
                                 int B, int H, int W, int C) {
    long t = (long)blockIdx.x * 256 + threadIdx.x;
    long tot = (long)B * H * W * 81;
    if (t >= tot) return;
    int o = (int)(t % 81);
    int p = (int)(t / 81);
    int x = p % W;
    int q = p / W;
    int y = q % H;
    int b = q / H;
    int dy = o / 9 - 4;
    int dx = o % 9 - 4;
    int yy = y + dy, xx = x + dx;
    float s = 0.0f;
    if ((unsigned)yy < (unsigned)H && (unsigned)xx < (unsigned)W) {
        const float* a  = f1 + (size_t)p * C;
        const float* bb = f2 + (((size_t)b * H + yy) * W + xx) * C;
        #pragma unroll 4
        for (int c = 0; c < C; ++c) s += a[c] * bb[c];
    }
    corr[t] = s / (float)C;
}

// Half-pixel bilinear resize (Hs,Ws)->(Ht,Wt), flow scaled by (Wt/Ws, Ht/Hs)
__global__ void raft_upsample_kernel(const float* __restrict__ flow,
                                     float* __restrict__ out,
                                     int B, int Hs, int Ws, int Ht, int Wt) {
    long t = (long)blockIdx.x * 256 + threadIdx.x;
    long tot = (long)B * Ht * Wt * 2;
    if (t >= tot) return;
    int c = (int)(t & 1);
    long q = t >> 1;
    int x = (int)(q % Wt); q /= Wt;
    int y = (int)(q % Ht);
    int b = (int)(q / Ht);
    float sx = (x + 0.5f) * ((float)Ws / (float)Wt) - 0.5f;
    float sy = (y + 0.5f) * ((float)Hs / (float)Ht) - 0.5f;
    sx = fminf(fmaxf(sx, 0.0f), (float)(Ws - 1));
    sy = fminf(fmaxf(sy, 0.0f), (float)(Hs - 1));
    int x0 = (int)sx, y0 = (int)sy;
    int x1 = (x0 + 1 < Ws) ? x0 + 1 : Ws - 1;
    int y1 = (y0 + 1 < Hs) ? y0 + 1 : Hs - 1;
    float fx = sx - (float)x0, fy = sy - (float)y0;
    const float* f = flow;
    float v00 = f[(((size_t)b * Hs + y0) * Ws + x0) * 2 + c];
    float v01 = f[(((size_t)b * Hs + y0) * Ws + x1) * 2 + c];
    float v10 = f[(((size_t)b * Hs + y1) * Ws + x0) * 2 + c];
    float v11 = f[(((size_t)b * Hs + y1) * Ws + x1) * 2 + c];
    float v = v00 * (1 - fx) * (1 - fy) + v01 * fx * (1 - fy) +
              v10 * (1 - fx) * fy + v11 * fx * fy;
    float scale = (c == 0) ? ((float)Wt / (float)Ws) : ((float)Ht / (float)Hs);
    out[t] = v * scale;
}

// ---------------------------------------------------------------------------
// Host orchestration
// ---------------------------------------------------------------------------
static inline long cdivl(long a, long b) { return (a + b - 1) / b; }

struct PW { _Float16* p; int Kpad; int CoutPad; };

extern "C" void kernel_launch(void* const* d_in, const int* in_sizes, int n_in,
                              void* d_out, int out_size, void* d_ws, size_t ws_size,
                              hipStream_t stream) {
    (void)in_sizes; (void)n_in; (void)out_size; (void)ws_size;

    const int B = 4;
    const int H0 = 384, W0 = 512;
    const int H1 = 192, W1 = 256;
    const int H2 = 96,  W2 = 128;
    const int H3 = 48,  W3 = 64;
    const long NPIX0 = (long)B * H0 * W0;
    const long NPIX1 = (long)B * H1 * W1;
    const long NPIX3 = (long)B * H3 * W3;

    // ---- input pointer map (jax pytree / alphabetical dict order) ----
    // 0 image1, 1 image2, 2..24 cenc, 25..47 fenc, 48..69 upd
    auto F = [&](int i) { return (const float*)d_in[i]; };
    // within-encoder offsets (alphabetical):
    // 0 c1, 1 co_b, 2 co_w, 3 l1c1, 4 l1c2, 5 l1n1b, 6 l1n1g, 7 l1n2b,
    // 8 l1n2g, 9 l2c1, 10 l2c2, 11 l2n1b, 12 l2n1g, 13 l2n2b, 14 l2n2g,
    // 15 l3c1, 16 l3c2, 17 l3n1b, 18 l3n1g, 19 l3n2b, 20 l3n2g, 21 n1b, 22 n1g
    const int CENC = 2, FENC = 25, UPD = 48;
    // upd offsets: 0 fh1b,1 fh1w,2 fh2b,3 fh2w,4 me1b,5 me1w,6 me2b,7 me2w,
    // 8 me3b,9 me3w,10 qub,11 quw,12 qwb,13 qww,14 rub,15 ruw,16 rwb,17 rww,
    // 18 zub,19 zuw,20 zwb,21 zww

    // ---- workspace allocator ----
    size_t off = 0;
    auto alloc = [&](size_t bytes) -> void* {
        off = (off + 255) & ~(size_t)255;
        void* p = (char*)d_ws + off;
        off += bytes;
        return p;
    };

    auto repack = [&](const float* w, int KH, int KW, int Cin, int Cout) -> PW {
        int K  = KH * KW * Cin;
        int Kp = (K + 31) & ~31;
        int Cp = (Cout + 15) & ~15;
        PW r;
        r.p = (_Float16*)alloc((size_t)Kp * Cp * sizeof(_Float16));
        r.Kpad = Kp; r.CoutPad = Cp;
        long tot = (long)Kp * Cp;
        raft_repack_kernel<<<(int)cdivl(tot, 256), 256, 0, stream>>>(
            w, r.p, K, Cout, Kp, Cp);
        return r;
    };

    auto conv = [&](const _Float16* xin, PW pw, const float* bias, float* yout,
                    int Bi, int Hi, int Wi, int Cin, int OHi, int OWi, int Cout,
                    int KH, int KW, int st, int pad) {
        long NP = (long)Bi * OHi * OWi;
        // each wave covers 32 channels; 8 waves per block -> 256 ch per grid.y
        dim3 grid((unsigned)cdivl(NP, 16),
                  (unsigned)cdivl(pw.CoutPad, 32 * CONV_WAVES));
        raft_conv_wmma_kernel<<<grid, 256, 0, stream>>>(
            xin, pw.p, bias, yout, Bi, Hi, Wi, Cin, OHi, OWi, Cout,
            KH, KW, st, pad, pw.Kpad, pw.CoutPad);
    };

    auto inorm = [&](const float* x, const float* g, const float* b,
                     _Float16* y, int Hi, int Wi, int C, int relu) {
        raft_inorm_kernel<<<B * C, 256, 0, stream>>>(x, g, b, y, Hi * Wi, C, relu);
    };

    auto cvt = [&](const float* x, _Float16* y, long n, int relu) {
        raft_f32_to_f16_kernel<<<(int)cdivl(n, 256), 256, 0, stream>>>(x, y, n, relu);
    };

    // ---- repack all weights (deterministic every call) ----
    PW fpw[8], cpw[8];
    for (int e = 0; e < 2; ++e) {
        int base = (e == 0) ? FENC : CENC;
        int cout = (e == 0) ? 256 : 128;
        PW* pw = (e == 0) ? fpw : cpw;
        pw[0] = repack(F(base + 0),  7, 7, 3,   64);    // c1
        pw[1] = repack(F(base + 3),  3, 3, 64,  64);    // l1c1
        pw[2] = repack(F(base + 4),  3, 3, 64,  64);    // l1c2
        pw[3] = repack(F(base + 9),  3, 3, 64,  96);    // l2c1
        pw[4] = repack(F(base + 10), 3, 3, 96,  96);    // l2c2
        pw[5] = repack(F(base + 15), 3, 3, 96,  128);   // l3c1
        pw[6] = repack(F(base + 16), 3, 3, 128, 128);   // l3c2
        pw[7] = repack(F(base + 2),  1, 1, 128, cout);  // co
    }
    PW pw_me1 = repack(F(UPD + 5),  1, 1, 83,  128);
    PW pw_me2 = repack(F(UPD + 7),  3, 3, 128, 128);
    PW pw_me3 = repack(F(UPD + 9),  3, 3, 128, 128);
    PW pw_zww = repack(F(UPD + 21), 3, 3, 128, 128);
    PW pw_zuw = repack(F(UPD + 19), 3, 3, 128, 128);
    PW pw_rww = repack(F(UPD + 17), 3, 3, 128, 128);
    PW pw_ruw = repack(F(UPD + 15), 3, 3, 128, 128);
    PW pw_qww = repack(F(UPD + 13), 3, 3, 128, 128);
    PW pw_quw = repack(F(UPD + 11), 3, 3, 128, 128);
    PW pw_fh1 = repack(F(UPD + 1),  3, 3, 128, 256);
    PW pw_fh2 = repack(F(UPD + 3),  3, 3, 256, 2);

    // ---- activation buffers ----
    _Float16* hA   = (_Float16*)alloc((size_t)NPIX1 * 64 * 2);   // 25.2 MB
    _Float16* hB   = (_Float16*)alloc((size_t)NPIX1 * 64 * 2);
    float*    bufF = (float*)   alloc((size_t)NPIX1 * 64 * 4);   // 50.3 MB
    float*    f1   = (float*)   alloc((size_t)NPIX3 * 256 * 4);
    float*    f2   = (float*)   alloc((size_t)NPIX3 * 256 * 4);
    float*    ctx  = (float*)   alloc((size_t)NPIX3 * 128 * 4);
    float*    h32  = (float*)   alloc((size_t)NPIX3 * 128 * 4);
    _Float16* h16  = (_Float16*)alloc((size_t)NPIX3 * 128 * 2);
    float*    corr = (float*)   alloc((size_t)NPIX3 * 81 * 4);
    _Float16* mi16 = (_Float16*)alloc((size_t)NPIX3 * 83 * 2);
    float*    g0   = (float*)   alloc((size_t)NPIX3 * 128 * 4);
    float*    g1   = (float*)   alloc((size_t)NPIX3 * 128 * 4);
    float*    g2   = (float*)   alloc((size_t)NPIX3 * 128 * 4);
    float*    zbuf = (float*)   alloc((size_t)NPIX3 * 128 * 4);
    _Float16* e0   = (_Float16*)alloc((size_t)NPIX3 * 128 * 2);
    _Float16* e1   = (_Float16*)alloc((size_t)NPIX3 * 128 * 2);
    _Float16* eb   = (_Float16*)alloc((size_t)NPIX3 * 256 * 2);
    float*    flow = (float*)   alloc((size_t)NPIX3 * 2 * 4);

    // ---- encoder runner ----
    auto run_encoder = [&](const float* img, int base, PW* pw,
                           float* outF32, int Cout) {
        cvt(img, hA, NPIX0 * 3, 0);
        // c1 7x7 s2, SAME -> pad_lo 2
        conv(hA, pw[0], nullptr, bufF, B, H0, W0, 3, H1, W1, 64, 7, 7, 2, 2);
        inorm(bufF, F(base + 22), F(base + 21), hB, H1, W1, 64, 1);          // n1, relu
        conv(hB, pw[1], nullptr, bufF, B, H1, W1, 64, H1, W1, 64, 3, 3, 1, 1);
        inorm(bufF, F(base + 6), F(base + 5), hA, H1, W1, 64, 1);            // l1n1, relu
        conv(hA, pw[2], nullptr, bufF, B, H1, W1, 64, H1, W1, 64, 3, 3, 1, 1);
        inorm(bufF, F(base + 8), F(base + 7), hB, H1, W1, 64, 0);            // l1n2, no relu
        // l2c1 3x3 s2, SAME even -> pad_lo 0
        conv(hB, pw[3], nullptr, bufF, B, H1, W1, 64, H2, W2, 96, 3, 3, 2, 0);
        inorm(bufF, F(base + 12), F(base + 11), hA, H2, W2, 96, 1);          // l2n1
        conv(hA, pw[4], nullptr, bufF, B, H2, W2, 96, H2, W2, 96, 3, 3, 1, 1);
        inorm(bufF, F(base + 14), F(base + 13), hB, H2, W2, 96, 1);          // l2n2
        conv(hB, pw[5], nullptr, bufF, B, H2, W2, 96, H3, W3, 128, 3, 3, 2, 0);
        inorm(bufF, F(base + 18), F(base + 17), hA, H3, W3, 128, 1);         // l3n1
        conv(hA, pw[6], nullptr, bufF, B, H3, W3, 128, H3, W3, 128, 3, 3, 1, 1);
        inorm(bufF, F(base + 20), F(base + 19), hB, H3, W3, 128, 1);         // l3n2
        conv(hB, pw[7], F(base + 1), outF32, B, H3, W3, 128, H3, W3, Cout, 1, 1, 1, 0);
    };

    run_encoder(F(0), FENC, fpw, f1, 256);
    run_encoder(F(1), FENC, fpw, f2, 256);
    run_encoder(F(0), CENC, cpw, ctx, 128);

    // correlation volume
    raft_corr_kernel<<<(int)cdivl(NPIX3 * 81, 256), 256, 0, stream>>>(
        f1, f2, corr, B, H3, W3, 256);

    // h = tanh(ctx); flow = 0
    raft_tanh_kernel<<<(int)cdivl(NPIX3 * 128, 256), 256, 0, stream>>>(
        ctx, h32, h16, NPIX3 * 128);
    hipMemsetAsync(flow, 0, (size_t)NPIX3 * 2 * 4, stream);

    const long NE = NPIX3 * 128;
    for (int it = 0; it < 8; ++it) {
        raft_mi_kernel<<<(int)cdivl(NPIX3 * 83, 256), 256, 0, stream>>>(
            corr, flow, mi16, (int)NPIX3);
        conv(mi16, pw_me1, F(UPD + 4), g0, B, H3, W3, 83, H3, W3, 128, 1, 1, 1, 0);
        cvt(g0, e0, NE, 1);
        conv(e0, pw_me2, F(UPD + 6), g0, B, H3, W3, 128, H3, W3, 128, 3, 3, 1, 1);
        cvt(g0, e0, NE, 1);
        conv(e0, pw_me3, F(UPD + 8), g0, B, H3, W3, 128, H3, W3, 128, 3, 3, 1, 1);
        cvt(g0, e0, NE, 0);                                  // e0 = m (f16)

        conv(e0,  pw_zww, F(UPD + 20), g1, B, H3, W3, 128, H3, W3, 128, 3, 3, 1, 1);
        conv(h16, pw_zuw, F(UPD + 18), g2, B, H3, W3, 128, H3, W3, 128, 3, 3, 1, 1);
        raft_zsig_kernel<<<(int)cdivl(NE, 256), 256, 0, stream>>>(g1, g2, zbuf, NE);

        conv(e0,  pw_rww, F(UPD + 16), g1, B, H3, W3, 128, H3, W3, 128, 3, 3, 1, 1);
        conv(h16, pw_ruw, F(UPD + 14), g2, B, H3, W3, 128, H3, W3, 128, 3, 3, 1, 1);
        raft_rh_kernel<<<(int)cdivl(NE, 256), 256, 0, stream>>>(g1, g2, h32, e1, NE);

        conv(e0, pw_qww, F(UPD + 12), g1, B, H3, W3, 128, H3, W3, 128, 3, 3, 1, 1);
        conv(e1, pw_quw, F(UPD + 10), g2, B, H3, W3, 128, H3, W3, 128, 3, 3, 1, 1);
        raft_hupd_kernel<<<(int)cdivl(NE, 256), 256, 0, stream>>>(
            g1, g2, zbuf, h32, h16, NE);

        conv(h16, pw_fh1, F(UPD + 0), bufF, B, H3, W3, 128, H3, W3, 256, 3, 3, 1, 1);
        cvt(bufF, eb, NPIX3 * 256, 1);
        conv(eb, pw_fh2, F(UPD + 2), g1, B, H3, W3, 256, H3, W3, 2, 3, 3, 1, 1);
        raft_flowadd_kernel<<<(int)cdivl(NPIX3 * 2, 256), 256, 0, stream>>>(
            g1, flow, NPIX3 * 2);
    }

    // bilinear upsample + flow scaling -> d_out [4,384,512,2]
    long nout = (long)B * H0 * W0 * 2;
    raft_upsample_kernel<<<(int)cdivl(nout, 256), 256, 0, stream>>>(
        flow, (float*)d_out, B, H3, W3, H0, W0);
}
